// JetBlockAttention_13683765805805
// MI455X (gfx1250) — matrix operand order repaired and checked by reading, not compiled, not run on hardware
//
#include <hip/hip_runtime.h>
#include <math.h>

typedef __bf16 bf16;
typedef __attribute__((ext_vector_type(16))) __bf16 v16bf;
typedef __attribute__((ext_vector_type(8)))  __bf16 v8bf;
typedef __attribute__((ext_vector_type(8)))  float   v8f;

// Problem constants
#define B_   4
#define T_   4096
#define MR_  16384      // B*T
#define HID_ 1024
#define NH_  16
#define QK_  64
#define VD_  64
#define KS_  4
#define RED_ 128

// ---------------- elementwise conversions ----------------

__global__ void k_cvt_f32_bf16(const float* __restrict__ in, bf16* __restrict__ out, long n) {
    long i = (long)blockIdx.x * blockDim.x + threadIdx.x;
    if (i < n) out[i] = (bf16)in[i];
}

// in: (K,N) row-major f32  ->  out: (N,K) row-major bf16
__global__ void k_transpose_cvt(const float* __restrict__ in, bf16* __restrict__ out, int K, int N) {
    long i = (long)blockIdx.x * blockDim.x + threadIdx.x;
    long total = (long)K * N;
    if (i >= total) return;
    int n = (int)(i / K);
    int k = (int)(i % K);
    out[i] = (bf16)in[(long)k * N + n];
}

// ---------------- CDNA5 async copy: global(16B) -> LDS, tracked by ASYNCcnt ----------------
__device__ __forceinline__ void async_copy16(unsigned lds_off, const bf16* gptr) {
    asm volatile("global_load_async_to_lds_b128 %0, %1, off"
                 :: "v"(lds_off), "v"((unsigned long long)(uintptr_t)gptr)
                 : "memory");
}

// ---------------- bf16 WMMA GEMM: C(MxN) = A(MxK) * B(KxN) + bias ----------------
// A row-major bf16, BT is B pre-transposed (NxK row-major bf16).
// mode 0: store f32 to outF;  mode 1: silu epilogue, store bf16 to outB.
// Double-buffered LDS staged with GLOBAL_LOAD_ASYNC_TO_LDS_B128; K-loop manually
// unrolled x2 so both buffers are statically addressed. Requires Kdim % 64 == 0.

#define BM 128
#define BN 128
#define BK 32
#define LDS_STRIDE 40   // 32 payload + 8 pad (bf16 units); 80B rows keep 16B alignment

__global__ __launch_bounds__(256)
void k_gemm_bf16(const bf16* __restrict__ A, const bf16* __restrict__ BT,
                 const float* __restrict__ bias,
                 float* __restrict__ outF, bf16* __restrict__ outB,
                 int Mrows, int Ncols, int Kdim, int mode)
{
    __shared__ bf16 As[2][BM * LDS_STRIDE];
    __shared__ bf16 Bs[2][BN * LDS_STRIDE];

    const int tid  = threadIdx.x;
    const int lane = tid & 31;
    const int wid  = tid >> 5;      // 0..7
    const int wr   = wid >> 1;      // 0..3 : row group of 32
    const int wc   = wid & 1;       // 0..1 : col group of 64
    const int m0   = blockIdx.y * BM;
    const int n0   = blockIdx.x * BN;
    const int lh   = lane >> 4;     // half-wave id
    const int ll   = lane & 15;
    (void)Mrows;

    // ---- per-thread staging chunks: 2 x 16B for A, 2 x 16B for B per K-step ----
    const bf16* gA[2];
    const bf16* gB[2];
    unsigned    ldsA[2][2];   // [it][buf]
    unsigned    ldsB[2][2];
    #pragma unroll
    for (int it = 0; it < 2; ++it) {
        int c    = tid + it * 256;
        int row  = c >> 2;
        int coff = (c & 3) * 8;
        gA[it] = A + (size_t)(m0 + row) * Kdim + coff;
        int nrow = n0 + row;
        if (nrow >= Ncols) nrow = Ncols - 1;   // clamp: junk cols >= N are never stored
        gB[it] = BT + (size_t)nrow * Kdim + coff;
        #pragma unroll
        for (int bfi = 0; bfi < 2; ++bfi) {
            ldsA[it][bfi] = (unsigned)(uintptr_t)&As[bfi][row * LDS_STRIDE + coff];
            ldsB[it][bfi] = (unsigned)(uintptr_t)&Bs[bfi][row * LDS_STRIDE + coff];
        }
    }

    auto issue_tile = [&](int kt, int bfi) {
        #pragma unroll
        for (int it = 0; it < 2; ++it) {
            async_copy16(ldsA[it][bfi], gA[it] + kt);
            async_copy16(ldsB[it][bfi], gB[it] + kt);
        }
    };

    v8f acc[2][4];
    #pragma unroll
    for (int i = 0; i < 2; ++i)
        #pragma unroll
        for (int j = 0; j < 4; ++j)
            #pragma unroll
            for (int r = 0; r < 8; ++r) acc[i][j][r] = 0.f;

    // 8 WMMAs on one statically-addressed buffer
    auto do_tiles = [&](const bf16* as, const bf16* bs) {
        v16bf af[2];
        #pragma unroll
        for (int i = 0; i < 2; ++i) {
            int mrow = wr * 32 + i * 16 + ll;
            const bf16* p = as + mrow * LDS_STRIDE + lh * 8;
            v8bf lo = *reinterpret_cast<const v8bf*>(p);
            v8bf hi = *reinterpret_cast<const v8bf*>(p + 16);
            #pragma unroll
            for (int q = 0; q < 8; ++q) { af[i][q] = lo[q]; af[i][8 + q] = hi[q]; }
        }
        #pragma unroll
        for (int j = 0; j < 4; ++j) {
            int nrow = wc * 64 + j * 16 + ll;
            const bf16* p = bs + nrow * LDS_STRIDE + lh * 8;
            v8bf lo = *reinterpret_cast<const v8bf*>(p);
            v8bf hi = *reinterpret_cast<const v8bf*>(p + 16);
            v16bf bb;
            #pragma unroll
            for (int q = 0; q < 8; ++q) { bb[q] = lo[q]; bb[8 + q] = hi[q]; }
            #pragma unroll
            for (int i = 0; i < 2; ++i) {
                acc[i][j] = __builtin_amdgcn_wmma_f32_16x16x32_bf16(
                    false, af[i], false, bb, (short)0, acc[i][j], false, false);
            }
        }
    };

    issue_tile(0, 0);

    // main loop: two ping-pong steps per iteration, all waits/addresses static
    int kt = 0;
    #pragma unroll 1
    for (; kt + 2 * BK < Kdim; kt += 2 * BK) {
        issue_tile(kt + BK, 1);                             // prefetch -> buf1
        asm volatile("s_wait_asynccnt 0x4" ::: "memory");   // buf0 landed (in-order)
        __syncthreads();
        do_tiles(As[0], Bs[0]);
        __syncthreads();

        issue_tile(kt + 2 * BK, 0);                         // prefetch -> buf0
        asm volatile("s_wait_asynccnt 0x4" ::: "memory");   // buf1 landed
        __syncthreads();
        do_tiles(As[1], Bs[1]);
        __syncthreads();
    }
    // tail pair (kt == Kdim - 64)
    issue_tile(kt + BK, 1);
    asm volatile("s_wait_asynccnt 0x4" ::: "memory");
    __syncthreads();
    do_tiles(As[0], Bs[0]);
    __syncthreads();

    asm volatile("s_wait_asynccnt 0x0" ::: "memory");
    __syncthreads();
    do_tiles(As[1], Bs[1]);
    // no trailing barrier needed: only global stores follow

    // ---- epilogue: D rows = r + 8*lh, col = ll within each 16x16 tile ----
    #pragma unroll
    for (int i = 0; i < 2; ++i) {
        #pragma unroll
        for (int j = 0; j < 4; ++j) {
            int nn = n0 + wc * 64 + j * 16 + ll;
            if (nn >= Ncols) continue;
            float bv = bias ? bias[nn] : 0.f;
            int mb = m0 + wr * 32 + i * 16 + lh * 8;
            #pragma unroll
            for (int r = 0; r < 8; ++r) {
                float d = acc[i][j][r] + bv;
                size_t off = (size_t)(mb + r) * Ncols + nn;
                if (mode == 1) {
                    float s = d / (1.f + __expf(-d));   // silu
                    outB[off] = (bf16)s;
                } else {
                    outF[off] = d;
                }
            }
        }
    }
}

// ---------------- gate: RoPE(Q) @ gate_w + gate_b -> sigmoid -> u,v ----------------
// One wave per (token m, head h). Lane i holds Q[i] and Q[i+32].
__global__ __launch_bounds__(256)
void k_gate(const float* __restrict__ Q, const float* __restrict__ gw,
            const float* __restrict__ gb,
            float* __restrict__ u, float* __restrict__ v)
{
    int lane = threadIdx.x & 31;
    int wid  = threadIdx.x >> 5;
    long idx = (long)blockIdx.x * 8 + wid;     // (m*NH + h)
    int m = (int)(idx / NH_);
    int h = (int)(idx % NH_);
    int t = m % T_;

    const float* qp = Q + (size_t)m * (NH_ * QK_) + h * QK_;
    float q1 = qp[lane];
    float q2 = qp[lane + 32];

    float freq = powf(10000.f, -2.f * (float)lane / 32.f);
    float ang  = (float)t * freq;
    float s, c;
    sincosf(ang, &s, &c);
    float rlo = q1 * c - q2 * s;
    float rhi = q1 * s + q2 * c;

    float a0 = rlo * gw[lane * 2 + 0] + rhi * gw[(lane + 32) * 2 + 0];
    float a1 = rlo * gw[lane * 2 + 1] + rhi * gw[(lane + 32) * 2 + 1];
    #pragma unroll
    for (int off = 16; off; off >>= 1) {
        a0 += __shfl_xor(a0, off);
        a1 += __shfl_xor(a1, off);
    }
    if (lane == 0) {
        u[(size_t)m * NH_ + h] = 1.f / (1.f + __expf(-(a0 + gb[0])));
        v[(size_t)m * NH_ + h] = 1.f / (1.f + __expf(-(a1 + gb[1])));
    }
}

// ---------------- Vmix (4-tap causal conv) + linear recurrence; emits bf16 ----------------
// One 64-thread block per (b,h); thread = VD channel d.
__global__ __launch_bounds__(64)
void k_scan(const float* __restrict__ V, const float* __restrict__ kern,
            const float* __restrict__ u, const float* __restrict__ v,
            bf16* __restrict__ attn)
{
    int bh = blockIdx.x;          // b*NH + h
    int b  = bh / NH_;
    int h  = bh % NH_;
    int d  = threadIdx.x;

    float ctx = 0.f, w1 = 0.f, w2 = 0.f, w3 = 0.f;
    for (int t = 0; t < T_; ++t) {
        size_t row = (size_t)(b * T_ + t);
        float vt = V[row * (NH_ * VD_) + h * VD_ + d];
        const float* kp = kern + row * (NH_ * KS_) + h * KS_;
        float vm = kp[3] * vt + kp[2] * w1 + kp[1] * w2 + kp[0] * w3;
        float uu = u[row * NH_ + h];
        float vv = v[row * NH_ + h];
        ctx = ctx * uu + vm * vv;
        attn[row * (NH_ * VD_) + h * VD_ + d] = (bf16)ctx;
        w3 = w2; w2 = w1; w1 = vt;
    }
}

// ---------------- launch ----------------

extern "C" void kernel_launch(void* const* d_in, const int* in_sizes, int n_in,
                              void* d_out, int out_size, void* d_ws, size_t ws_size,
                              hipStream_t stream) {
    (void)in_sizes; (void)n_in; (void)out_size; (void)ws_size;

    const float* x     = (const float*)d_in[0];
    const float* W_q   = (const float*)d_in[1];
    const float* b_q   = (const float*)d_in[2];
    // d_in[3]=W_k, d_in[4]=b_k are dead in the reference -> skipped
    const float* W_v   = (const float*)d_in[5];
    const float* b_v   = (const float*)d_in[6];
    const float* kg_w1 = (const float*)d_in[7];
    const float* kg_b1 = (const float*)d_in[8];
    const float* kg_w2 = (const float*)d_in[9];
    const float* kg_b2 = (const float*)d_in[10];
    const float* gw    = (const float*)d_in[11];
    const float* gb    = (const float*)d_in[12];
    const float* W_out = (const float*)d_in[13];
    const float* b_out = (const float*)d_in[14];
    float* out = (float*)d_out;

    char* ws = (char*)d_ws;
    size_t off = 0;
    auto alloc = [&](size_t bytes) -> char* {
        char* p = ws + off;
        off = (off + bytes + 255) & ~(size_t)255;
        return p;
    };
    bf16*  xb    = (bf16*)alloc((size_t)MR_ * HID_ * 2);
    bf16*  WqT   = (bf16*)alloc((size_t)HID_ * (NH_*QK_) * 2);
    bf16*  WvT   = (bf16*)alloc((size_t)HID_ * (NH_*VD_) * 2);
    bf16*  Wg1T  = (bf16*)alloc((size_t)HID_ * RED_ * 2);
    bf16*  Wg2T  = (bf16*)alloc((size_t)RED_ * (NH_*KS_) * 2);
    bf16*  WoT   = (bf16*)alloc((size_t)(NH_*VD_) * HID_ * 2);
    float* Qf    = (float*)alloc((size_t)MR_ * (NH_*QK_) * 4);
    float* Vf    = (float*)alloc((size_t)MR_ * (NH_*VD_) * 4);
    bf16*  Rb    = (bf16*)alloc((size_t)MR_ * RED_ * 2);
    float* kern  = (float*)alloc((size_t)MR_ * (NH_*KS_) * 4);
    float* uf    = (float*)alloc((size_t)MR_ * NH_ * 4);
    float* vf    = (float*)alloc((size_t)MR_ * NH_ * 4);
    bf16*  attnb = (bf16*)alloc((size_t)MR_ * (NH_*VD_) * 2);

    // 1) convert x to bf16
    {
        long n = (long)MR_ * HID_;
        k_cvt_f32_bf16<<<(unsigned)((n + 255) / 256), 256, 0, stream>>>(x, xb, n);
    }
    // 2) transpose+convert weights to bf16 (N x K layout)
    auto tcvt = [&](const float* in, bf16* o, int K, int N) {
        long n = (long)K * N;
        k_transpose_cvt<<<(unsigned)((n + 255) / 256), 256, 0, stream>>>(in, o, K, N);
    };
    tcvt(W_q,   WqT,  HID_, NH_*QK_);
    tcvt(W_v,   WvT,  HID_, NH_*VD_);
    tcvt(kg_w1, Wg1T, HID_, RED_);
    tcvt(kg_w2, Wg2T, RED_, NH_*KS_);
    tcvt(W_out, WoT,  NH_*VD_, HID_);

    // 3) GEMMs  (Kdim must be a multiple of 64: 1024, 1024, 1024, 128, 1024 -> ok)
    auto gemm = [&](const bf16* A, const bf16* BT, const float* bias,
                    float* oF, bf16* oB, int M, int N, int K, int mode) {
        dim3 grid((N + BN - 1) / BN, M / BM);
        k_gemm_bf16<<<grid, 256, 0, stream>>>(A, BT, bias, oF, oB, M, N, K, mode);
    };
    gemm(xb, WqT,  b_q,   Qf,   nullptr, MR_, NH_*QK_, HID_, 0);   // Q
    gemm(xb, WvT,  b_v,   Vf,   nullptr, MR_, NH_*VD_, HID_, 0);   // V
    gemm(xb, Wg1T, kg_b1, nullptr, Rb,   MR_, RED_,    HID_, 1);   // silu(x@kg_w1) -> bf16
    gemm(Rb, Wg2T, kg_b2, kern, nullptr, MR_, NH_*KS_, RED_, 0);   // kernels

    // 4) gates (RoPE fused)
    k_gate<<<(MR_ * NH_) / 8, 256, 0, stream>>>(Qf, gw, gb, uf, vf);

    // 5) conv + recurrence -> attn (bf16)
    k_scan<<<B_ * NH_, 64, 0, stream>>>(Vf, kern, uf, vf, attnb);

    // 6) final projection
    gemm(attnb, WoT, b_out, out, nullptr, MR_, HID_, NH_*VD_, 0);
}
